// VAE_17179869733
// MI455X (gfx1250) — matrix operand order repaired and checked by reading, not compiled, not run on hardware
//
#include <hip/hip_runtime.h>

// ---------------------------------------------------------------------------
// Problem constants (from reference)
// ---------------------------------------------------------------------------
#define ROLL   130
#define HID    1024
#define INF    256
#define NSTEP  64
#define BATCH  1024

#define KX1    416              // padded ROLL+INF (386 -> 416, multiple of 32)
#define NOUTP  144              // padded ROLL (130 -> 144, multiple of 16)

#define ROWS_PER_BLOCK 16
#define BLOCK_THREADS  512      // 16 waves
#define NWAVES         (BLOCK_THREADS / 32)

typedef __bf16 bf16_t;
typedef __attribute__((ext_vector_type(16))) __bf16 v16bf;
typedef __attribute__((ext_vector_type(8)))  __bf16 v8bf;
typedef __attribute__((ext_vector_type(8)))  float  v8f;

#define LOG2E   1.4426950408889634f
#define LN2     0.6931471805599453f

// ---------------------------------------------------------------------------
// Branchless fast transcendentals on native v_exp/v_log/v_rcp/v_tanh.
// ---------------------------------------------------------------------------
__device__ __forceinline__ float fast_sigmoid(float x) {
    return __builtin_amdgcn_rcpf(1.0f + __builtin_amdgcn_exp2f(-LOG2E * x));
}

__device__ __forceinline__ float fast_tanh(float x) {
#if __has_builtin(__builtin_amdgcn_tanhf)
    return __builtin_amdgcn_tanhf(x);
#else
    const float ax = __builtin_fabsf(x);
    const float t  = __builtin_amdgcn_exp2f(-2.0f * LOG2E * ax);
    const float y  = 1.0f - (2.0f * t) * __builtin_amdgcn_rcpf(1.0f + t);
    return __builtin_copysignf(y, x);
#endif
}

// ---------------------------------------------------------------------------
// Fragment loaders (CDNA5 16-bit WMMA layouts, wave32)
//
// A (16x32, MxK): lanes 0-15: M=lane, K={0..7} in v0..3, K={16..23} in v4..7
//                 lanes 16-31: M=lane-16, K={8..15} / {24..31}
// B (32x16, KxN): lanes 0-15: N=lane, K={0..7,16..23}; lanes 16-31: N=lane-16, K={8..15,24..31}
// C/D (16x16 f32): VGPR r, lane L: M = r + 8*(L>>4), N = L&15
// ---------------------------------------------------------------------------
__device__ __forceinline__ v16bf load_a_frag(const bf16_t* base, int ld, int k0, int lane) {
    const int m  = lane & 15;
    const int hi = lane >> 4;
    const bf16_t* p = base + (size_t)m * ld + k0 + hi * 8;
    v8bf lo = *(const v8bf*)(p);
    v8bf hh = *(const v8bf*)(p + 16);
    v16bf a;
#pragma unroll
    for (int i = 0; i < 8; ++i) { a[i] = lo[i]; a[i + 8] = hh[i]; }
    return a;
}

__device__ __forceinline__ v16bf load_b_frag(const bf16_t* base, int ld, int k0, int lane) {
    const int n  = lane & 15;
    const int hi = lane >> 4;
    const bf16_t* p = base + (size_t)n * ld + k0 + hi * 8;
    v8bf b0 = *(const v8bf*)(p);
    v8bf b1 = *(const v8bf*)(p + 16);
    v16bf b;
#pragma unroll
    for (int i = 0; i < 8; ++i) { b[i] = b0[i]; b[i + 8] = b1[i]; }
    return b;
}

#define WMMA_BF16(a, b, c) \
    __builtin_amdgcn_wmma_f32_16x16x32_bf16(false, (a), false, (b), (short)0, (c), false, false)

// acc += X[16 x K] @ W[16 x K]^T   (single accumulator; init/output GEMMs)
__device__ __forceinline__ v8f gemm_acc(v8f acc, const bf16_t* X, int ldx, int K,
                                        const bf16_t* W, int ldw, int lane) {
#pragma unroll 4
    for (int k0 = 0; k0 < K; k0 += 32) {
        v16bf a = load_a_frag(X, ldx, k0, lane);
        v16bf b = load_b_frag(W, ldw, k0, lane);
        acc = WMMA_BF16(a, b, acc);
    }
    return acc;
}

// ---------------------------------------------------------------------------
// One GRU cell for this block's 16 rows. Each wave owns 4 n-tiles, processed
// as 2 fused pairs: one A-frag LDS load feeds SIX WMMAs (2 tiles x 3 gates).
// X: [16][Kx] bf16 LDS, Hprev/Hnew: [16][HID] bf16 LDS.
// Wih: [3*HID][Kx] bf16, Whh: [3*HID][HID] bf16 (global, L2-resident).
// ---------------------------------------------------------------------------
__device__ __forceinline__ void gru_wave(const bf16_t* X, int Kx,
                                         const bf16_t* __restrict__ Wih,
                                         const bf16_t* __restrict__ Whh,
                                         const float* __restrict__ b_ih,
                                         const float* __restrict__ b_hh,
                                         const bf16_t* Hprev, bf16_t* Hnew,
                                         int wave, int lane) {
    const int tiles_per_wave = (HID / 16) / NWAVES;  // 4
#pragma unroll 1
    for (int jp = 0; jp < tiles_per_wave; jp += 2) {
        const int nb = (wave * tiles_per_wave + jp) * 16;
        v8f ar[2]  = {{}, {}};
        v8f az[2]  = {{}, {}};
        v8f ani[2] = {{}, {}};
        v8f anh[2] = {{}, {}};

        const bf16_t* WiR = Wih + (size_t)(0 * HID + nb) * Kx;
        const bf16_t* WiZ = Wih + (size_t)(1 * HID + nb) * Kx;
        const bf16_t* WiN = Wih + (size_t)(2 * HID + nb) * Kx;
#pragma unroll 2
        for (int k0 = 0; k0 < Kx; k0 += 32) {
            v16bf a = load_a_frag(X, Kx, k0, lane);
            ar[0]  = WMMA_BF16(a, load_b_frag(WiR,                  Kx, k0, lane), ar[0]);
            az[0]  = WMMA_BF16(a, load_b_frag(WiZ,                  Kx, k0, lane), az[0]);
            ani[0] = WMMA_BF16(a, load_b_frag(WiN,                  Kx, k0, lane), ani[0]);
            ar[1]  = WMMA_BF16(a, load_b_frag(WiR + (size_t)16 * Kx, Kx, k0, lane), ar[1]);
            az[1]  = WMMA_BF16(a, load_b_frag(WiZ + (size_t)16 * Kx, Kx, k0, lane), az[1]);
            ani[1] = WMMA_BF16(a, load_b_frag(WiN + (size_t)16 * Kx, Kx, k0, lane), ani[1]);
        }

        const bf16_t* WhR = Whh + (size_t)(0 * HID + nb) * HID;
        const bf16_t* WhZ = Whh + (size_t)(1 * HID + nb) * HID;
        const bf16_t* WhN = Whh + (size_t)(2 * HID + nb) * HID;
#pragma unroll 2
        for (int k0 = 0; k0 < HID; k0 += 32) {
            v16bf a = load_a_frag(Hprev, HID, k0, lane);
            ar[0]  = WMMA_BF16(a, load_b_frag(WhR,                   HID, k0, lane), ar[0]);
            az[0]  = WMMA_BF16(a, load_b_frag(WhZ,                   HID, k0, lane), az[0]);
            anh[0] = WMMA_BF16(a, load_b_frag(WhN,                   HID, k0, lane), anh[0]);
            ar[1]  = WMMA_BF16(a, load_b_frag(WhR + (size_t)16 * HID, HID, k0, lane), ar[1]);
            az[1]  = WMMA_BF16(a, load_b_frag(WhZ + (size_t)16 * HID, HID, k0, lane), az[1]);
            anh[1] = WMMA_BF16(a, load_b_frag(WhN + (size_t)16 * HID, HID, k0, lane), anh[1]);
        }

        const int hi = lane >> 4;
#pragma unroll
        for (int q = 0; q < 2; ++q) {
            const int n   = nb + q * 16 + (lane & 15);
            const float br  = b_ih[n]           + b_hh[n];
            const float bz  = b_ih[HID + n]     + b_hh[HID + n];
            const float bin = b_ih[2 * HID + n];
            const float bhn = b_hh[2 * HID + n];
#pragma unroll
            for (int i = 0; i < 8; ++i) {
                const int m = i + hi * 8;
                const float r  = fast_sigmoid(ar[q][i] + br);
                const float z  = fast_sigmoid(az[q][i] + bz);
                const float nn = fast_tanh(ani[q][i] + bin + r * (anh[q][i] + bhn));
                const float hp = (float)Hprev[(size_t)m * HID + n];
                Hnew[(size_t)m * HID + n] = (bf16_t)((1.0f - z) * nn + z * hp);
            }
        }
    }
}

// ---------------------------------------------------------------------------
// Weight prep: f32 -> bf16, pad K to 32-multiples, permute w_ih1 columns to
// [infor | onehot | zero-pad], pad w_out rows 130->144 with zeros.
// ---------------------------------------------------------------------------
__global__ void prep_weights(const float* __restrict__ wih1, const float* __restrict__ whh1,
                             const float* __restrict__ wih2, const float* __restrict__ whh2,
                             const float* __restrict__ winit, const float* __restrict__ wout,
                             bf16_t* __restrict__ Wih1p, bf16_t* __restrict__ Whh1,
                             bf16_t* __restrict__ Wih2,  bf16_t* __restrict__ Whh2,
                             bf16_t* __restrict__ Winit, bf16_t* __restrict__ Woutp) {
    const size_t tid    = (size_t)blockIdx.x * blockDim.x + threadIdx.x;
    const size_t stride = (size_t)gridDim.x * blockDim.x;

    for (size_t i = tid; i < (size_t)3 * HID * KX1; i += stride) {
        const int n = (int)(i / KX1), c = (int)(i % KX1);
        float v = 0.0f;
        if (c < INF)                 v = wih1[(size_t)n * (ROLL + INF) + ROLL + c];
        else if (c < INF + ROLL)     v = wih1[(size_t)n * (ROLL + INF) + (c - INF)];
        Wih1p[i] = (bf16_t)v;
    }
    for (size_t i = tid; i < (size_t)3 * HID * HID; i += stride) {
        Whh1[i] = (bf16_t)whh1[i];
        Wih2[i] = (bf16_t)wih2[i];
        Whh2[i] = (bf16_t)whh2[i];
    }
    for (size_t i = tid; i < (size_t)HID * INF; i += stride) Winit[i] = (bf16_t)winit[i];
    for (size_t i = tid; i < (size_t)NOUTP * HID; i += stride) {
        const int n = (int)(i / HID);
        Woutp[i] = (bf16_t)(n < ROLL ? wout[i] : 0.0f);
    }
}

// ---------------------------------------------------------------------------
// Persistent GRU sampler: 1 block = 16 batch rows, all 64 steps.
// LDS: h1 x2, h2 x2 ([16][1024] bf16), xin [16][416] bf16, logits [16][144] f32.
// ---------------------------------------------------------------------------
#define SMEM_BYTES (4 * ROWS_PER_BLOCK * HID * 2 + ROWS_PER_BLOCK * KX1 * 2 \
                    + ROWS_PER_BLOCK * NOUTP * 4 + 4 * ROWS_PER_BLOCK * 4)

__global__ __launch_bounds__(BLOCK_THREADS, 1)
void gru_sampler(const float* __restrict__ infor, const int* __restrict__ length,
                 const float* __restrict__ b_ih1, const float* __restrict__ b_hh1,
                 const float* __restrict__ b_ih2, const float* __restrict__ b_hh2,
                 const float* __restrict__ b_init, const float* __restrict__ b_out,
                 const bf16_t* __restrict__ Wih1p, const bf16_t* __restrict__ Whh1,
                 const bf16_t* __restrict__ Wih2,  const bf16_t* __restrict__ Whh2,
                 const bf16_t* __restrict__ Winit, const bf16_t* __restrict__ Woutp,
                 float* __restrict__ out) {
    extern __shared__ char smem[];
    bf16_t* h1a = (bf16_t*)smem;
    bf16_t* h1b = h1a + ROWS_PER_BLOCK * HID;
    bf16_t* h2a = h1b + ROWS_PER_BLOCK * HID;
    bf16_t* h2b = h2a + ROWS_PER_BLOCK * HID;
    bf16_t* xin = h2b + ROWS_PER_BLOCK * HID;              // [16][KX1]
    float*  lg  = (float*)(xin + ROWS_PER_BLOCK * KX1);    // [16][NOUTP]
    float*  rowmax = lg + ROWS_PER_BLOCK * NOUTP;
    float*  rowlse = rowmax + ROWS_PER_BLOCK;
    int*    sel    = (int*)(rowlse + ROWS_PER_BLOCK);
    int*    len    = sel + ROWS_PER_BLOCK;

    const int tid  = threadIdx.x;
    const int wave = tid >> 5;
    const int lane = tid & 31;
    const int b0   = blockIdx.x * ROWS_PER_BLOCK;

    // ---- init: infor slice -> xin[:,0:256] bf16, pad cols, lengths, out0 one-hot
    for (int i = tid; i < ROWS_PER_BLOCK * INF; i += BLOCK_THREADS) {
        const int m = i >> 8, c = i & 255;
        xin[m * KX1 + c] = (bf16_t)infor[(size_t)(b0 + m) * INF + c];
    }
    for (int i = tid; i < ROWS_PER_BLOCK * (KX1 - INF - ROLL); i += BLOCK_THREADS) {
        const int m = i / (KX1 - INF - ROLL), c = i % (KX1 - INF - ROLL);
        xin[m * KX1 + INF + ROLL + c] = (bf16_t)0.0f;
    }
    if (tid < ROWS_PER_BLOCK) {
        len[tid] = length[b0 + tid];
        sel[tid] = ROLL - 1;          // out0 one-hot at last index
    }
    __syncthreads();

    // ---- h1 = tanh(infor @ Winit^T + b_init)
    {
        const int tiles_per_wave = (HID / 16) / NWAVES;
#pragma unroll 1
        for (int j = 0; j < tiles_per_wave; ++j) {
            const int nb = (wave * tiles_per_wave + j) * 16;
            const int n  = nb + (lane & 15);
            v8f acc = {};
            acc = gemm_acc(acc, xin, KX1, INF, Winit + (size_t)nb * INF, INF, lane);
            const float bi = b_init[n];
            const int hi = lane >> 4;
#pragma unroll
            for (int i = 0; i < 8; ++i)
                h1a[(size_t)(i + hi * 8) * HID + n] = (bf16_t)fast_tanh(acc[i] + bi);
        }
    }

    bf16_t *h1c = h1a, *h1n = h1b, *h2c = h2a, *h2n = h2b;

#pragma unroll 1
    for (int t = 0; t < NSTEP; ++t) {
        __syncthreads();
        // build one-hot part of xin from previous argmax
        for (int i = tid; i < ROWS_PER_BLOCK * ROLL; i += BLOCK_THREADS) {
            const int m = i / ROLL, c = i - m * ROLL;
            xin[m * KX1 + INF + c] = (bf16_t)((c == sel[m]) ? 1.0f : 0.0f);
        }
        __syncthreads();

        // h1n = GRU1([out,infor], h1c)
        gru_wave(xin, KX1, Wih1p, Whh1, b_ih1, b_hh1, h1c, h1n, wave, lane);
        __syncthreads();

        // h2n = GRU2(h1n, t==0 ? h1n : h2c)
        gru_wave(h1n, HID, Wih2, Whh2, b_ih2, b_hh2, (t == 0) ? h1n : h2c, h2n, wave, lane);
        __syncthreads();

        // logits = h2n @ Woutp^T + b_out  (9 n-tiles over waves 0..8)
        if (wave < NOUTP / 16) {
            const int nb = wave * 16;
            const int n  = nb + (lane & 15);
            v8f acc = {};
            acc = gemm_acc(acc, h2n, HID, HID, Woutp + (size_t)nb * HID, HID, lane);
            const float bo = (n < ROLL) ? b_out[n] : -1e30f;
            const int hi = lane >> 4;
#pragma unroll
            for (int i = 0; i < 8; ++i)
                lg[(i + hi * 8) * NOUTP + n] = acc[i] + bo;
        }
        __syncthreads();

        // per-row max / argmax / logsumexp: one wave per row, lane-parallel,
        // wave32 butterfly reductions (first-index tie-breaking like argmax).
        {
            const int m = wave;          // 16 waves == 16 rows
            float mx = -1e30f; int am = ROLL;
            for (int c = lane; c < ROLL; c += 32) {
                const float v = lg[m * NOUTP + c];
                if (v > mx) { mx = v; am = c; }
            }
#pragma unroll
            for (int off = 16; off > 0; off >>= 1) {
                const float ov = __shfl_xor(mx, off, 32);
                const int   oi = __shfl_xor(am, off, 32);
                if (ov > mx || (ov == mx && oi < am)) { mx = ov; am = oi; }
            }
            float s = 0.0f;
            for (int c = lane; c < ROLL; c += 32)
                s += __builtin_amdgcn_exp2f((lg[m * NOUTP + c] - mx) * LOG2E);
#pragma unroll
            for (int off = 16; off > 0; off >>= 1) s += __shfl_xor(s, off, 32);
            if (lane == 0) {
                rowmax[m] = mx;
                rowlse[m] = __builtin_amdgcn_logf(s) * LN2;
                sel[m]    = am;
            }
        }
        __syncthreads();

        // write masked log-softmax output
        for (int i = tid; i < ROWS_PER_BLOCK * ROLL; i += BLOCK_THREADS) {
            const int m = i / ROLL, c = i - m * ROLL;
            float v;
            if (t >= len[m]) v = (c == ROLL - 1) ? -1.0f : 0.0f;
            else             v = lg[m * NOUTP + c] - rowmax[m] - rowlse[m];
            out[((size_t)(b0 + m) * NSTEP + t) * ROLL + c] = v;
        }

        // ping-pong hidden buffers
        bf16_t* tmp;
        tmp = h1c; h1c = h1n; h1n = tmp;
        tmp = h2c; h2c = h2n; h2n = tmp;
    }
}

// ---------------------------------------------------------------------------
extern "C" void kernel_launch(void* const* d_in, const int* in_sizes, int n_in,
                              void* d_out, int out_size, void* d_ws, size_t ws_size,
                              hipStream_t stream) {
    const float* infor  = (const float*)d_in[0];
    // d_in[1] ("x") is unused by the reference
    const int*   length = (const int*)d_in[2];
    const float* w_ih1  = (const float*)d_in[3];
    const float* w_hh1  = (const float*)d_in[4];
    const float* b_ih1  = (const float*)d_in[5];
    const float* b_hh1  = (const float*)d_in[6];
    const float* w_ih2  = (const float*)d_in[7];
    const float* w_hh2  = (const float*)d_in[8];
    const float* b_ih2  = (const float*)d_in[9];
    const float* b_hh2  = (const float*)d_in[10];
    const float* w_init = (const float*)d_in[11];
    const float* b_init = (const float*)d_in[12];
    const float* w_out  = (const float*)d_in[13];
    const float* b_out  = (const float*)d_in[14];
    float* out = (float*)d_out;

    char* p = (char*)d_ws;
    bf16_t* Wih1p = (bf16_t*)p; p += (size_t)3 * HID * KX1 * 2;
    bf16_t* Whh1  = (bf16_t*)p; p += (size_t)3 * HID * HID * 2;
    bf16_t* Wih2  = (bf16_t*)p; p += (size_t)3 * HID * HID * 2;
    bf16_t* Whh2  = (bf16_t*)p; p += (size_t)3 * HID * HID * 2;
    bf16_t* Winit = (bf16_t*)p; p += (size_t)HID * INF * 2;
    bf16_t* Woutp = (bf16_t*)p; p += (size_t)NOUTP * HID * 2;

    prep_weights<<<512, 256, 0, stream>>>(w_ih1, w_hh1, w_ih2, w_hh2, w_init, w_out,
                                          Wih1p, Whh1, Wih2, Whh2, Winit, Woutp);

    gru_sampler<<<BATCH / ROWS_PER_BLOCK, BLOCK_THREADS, SMEM_BYTES, stream>>>(
        infor, length, b_ih1, b_hh1, b_ih2, b_hh2, b_init, b_out,
        Wih1p, Whh1, Wih2, Whh2, Winit, Woutp, out);
}
